// TFN_25778393710947
// MI455X (gfx1250) — matrix operand order repaired and checked by reading, not compiled, not run on hardware
//
#include <hip/hip_runtime.h>

// ---------------------------------------------------------------------------
// Problem constants
// ---------------------------------------------------------------------------
#define NN 20000
#define EE 320000
#define CC 16
#define MIDD 32

// Workspace layout (floats). Required ws_size >= 3,981,352 * 4 B (~15.2 MiB).
#define OFF_UR   0            // E*4 : (ux,uy,uz,r) per edge
#define OFF_DEG  1280000      // N
#define OFF_H0A  1300000      // N*16
#define OFF_H1A  1620000      // N*48
#define OFF_OUTA 2580000      // N*6
#define OFF_S1   2700000      // 8   : feat moments
#define OFF_S2   2700008      // 384 : layer-2 stats (sum, sumsq)
#define OFF_BN1  2700392      // 576 : folded BN1 (wcoef, rcoef, bias) [3][6][32]
#define OFF_BN2  2700968      // 384 : folded BN2 (scale, shift) per (k,col)
#define OFF_H0N  2701352      // N*16
#define OFF_H1N  3021352      // N*48
#define ZERO_OFF OFF_DEG
#define ZERO_CNT (OFF_BN1 - OFF_DEG)   // deg..stats2 contiguous = 1,420,392

typedef __attribute__((ext_vector_type(16))) _Float16 v16h;
typedef __attribute__((ext_vector_type(8)))  _Float16 v8h;
typedef __attribute__((ext_vector_type(8)))  float    v8f;

// D = A(16x32 f16) * B(32x16 f16) + C(16x16 f32)
__device__ __forceinline__ v8f wmma_f16(v16h a, v16h b, v8f c) {
  return __builtin_amdgcn_wmma_f32_16x16x32_f16(false, a, false, b, (short)0, c,
                                                false, false);
}

// Load A-tile from LDS stage[16][32] (row-major f16) in contiguous K order
// (matching the B staging): lane<16 -> M=lane, K {0..15};
// lane>=16 -> M=lane-16, K {16..31}.
__device__ __forceinline__ v16h loadA(const _Float16* st, int lane) {
  const int e   = lane & 15;
  const int off = (lane >> 4) * 16;
  const _Float16* row = st + e * 32;
  v8h lo = *reinterpret_cast<const v8h*>(row + off);
  v8h hi = *reinterpret_cast<const v8h*>(row + off + 8);
  return __builtin_shufflevector(lo, hi, 0,1,2,3,4,5,6,7,8,9,10,11,12,13,14,15);
}

// Stage a 32xoutDim f32 weight matrix's 16-column tile into LDS B layout:
// dst[lane*16+i] = W[Kbase(lane)+i][colBase + (lane&15)], Kbase = 16*(lane>=16).
__device__ __forceinline__ void stageW3(_Float16* dstp, const float* w3,
                                        int outDim, int colBase, int tid) {
  for (int idx = tid; idx < 512; idx += 128) {
    int ln = idx >> 4, i = idx & 15;
    dstp[idx] = (_Float16)w3[((ln & 16) + i) * outDim + colBase + (ln & 15)];
  }
}

// ---------------------------------------------------------------------------
// K0: zero the accumulators / stats region of the workspace
// ---------------------------------------------------------------------------
__global__ void k_zero(float* p, int n) {
  int i = blockIdx.x * blockDim.x + threadIdx.x;
  if (i < n) p[i] = 0.f;
}

// ---------------------------------------------------------------------------
// K1: per-edge geometry, degree, feat moments (w, r, w^2, wr, r^2)
// ---------------------------------------------------------------------------
__global__ void k_geom(const int* __restrict__ src, const int* __restrict__ dst,
                       const float* __restrict__ pos, const float* __restrict__ w,
                       float* __restrict__ uR, float* __restrict__ deg,
                       float* __restrict__ stats1) {
  int e = blockIdx.x * blockDim.x + threadIdx.x;
  float wv = 0.f, rv = 0.f;
  if (e < EE) {
    int s = src[e], d = dst[e];
    float rx = pos[d * 3 + 0] - pos[s * 3 + 0];
    float ry = pos[d * 3 + 1] - pos[s * 3 + 1];
    float rz = pos[d * 3 + 2] - pos[s * 3 + 2];
    float r  = sqrtf(rx * rx + ry * ry + rz * rz);
    float ir = 1.f / fmaxf(r, 1e-8f);
    uR[e * 4 + 0] = rx * ir;
    uR[e * 4 + 1] = ry * ir;
    uR[e * 4 + 2] = rz * ir;
    uR[e * 4 + 3] = r;
    atomicAdd(&deg[d], 1.f);
    wv = w[e]; rv = r;
  }
  float s0 = wv, s1 = rv, s2 = wv * wv, s3 = wv * rv, s4 = rv * rv;
  for (int off = 16; off > 0; off >>= 1) {
    s0 += __shfl_xor(s0, off, 32); s1 += __shfl_xor(s1, off, 32);
    s2 += __shfl_xor(s2, off, 32); s3 += __shfl_xor(s3, off, 32);
    s4 += __shfl_xor(s4, off, 32);
  }
  if ((threadIdx.x & 31) == 0) {
    atomicAdd(&stats1[0], s0); atomicAdd(&stats1[1], s1);
    atomicAdd(&stats1[2], s2); atomicAdd(&stats1[3], s3);
    atomicAdd(&stats1[4], s4);
  }
}

// ---------------------------------------------------------------------------
// K2: fold BN1 analytically (input to BN1 is affine in the 2-dim feat)
// bn1 layout: [0]=w-coef, [1]=r-coef, [2]=bias, each [6][32]
// ---------------------------------------------------------------------------
__global__ void k_bn1(const float* __restrict__ stats1, const float* __restrict__ rW1,
                      const float* __restrict__ rb1, const float* __restrict__ rg1,
                      const float* __restrict__ rbe1, float* __restrict__ bn1) {
  int t = threadIdx.x;
  if (t >= 192) return;
  int k = t >> 5, j = t & 31;
  const float invE = 1.f / (float)EE;
  float mw  = stats1[0] * invE, mr = stats1[1] * invE;
  float cww = stats1[2] * invE - mw * mw;
  float cwr = stats1[3] * invE - mw * mr;
  float crr = stats1[4] * invE - mr * mr;
  float w0 = rW1[(k * 2 + 0) * 32 + j];
  float w1 = rW1[(k * 2 + 1) * 32 + j];
  float mu  = mw * w0 + mr * w1 + rb1[k * 32 + j];
  float var = cww * w0 * w0 + 2.f * cwr * w0 * w1 + crr * w1 * w1;
  float s   = rsqrtf(var + 1e-5f) * rg1[k * 32 + j];
  bn1[0 * 192 + k * 32 + j] = w0 * s;
  bn1[1 * 192 + k * 32 + j] = w1 * s;
  bn1[2 * 192 + k * 32 + j] = (rb1[k * 32 + j] - mu) * s + rbe1[k * 32 + j];
}

// ---------------------------------------------------------------------------
// K3 (WMMA): stats pass for BN2: y = h1 @ W2 + b2, accumulate sum(y), sum(y^2)
// 4 waves / 128 threads, 64 edges per block, 16 edges per wave.
// ---------------------------------------------------------------------------
__global__ __launch_bounds__(128) void k_stats2(
    const float* __restrict__ w, const float* __restrict__ uR,
    const float* __restrict__ bn1, const float* __restrict__ rW2,
    const float* __restrict__ rb2, float* __restrict__ stats2) {
  __shared__ alignas(32) _Float16 W2B[6 * 2 * 512];
  __shared__ alignas(32) _Float16 stage[4][16][32];
  __shared__ float ewr[4][16][2];
  __shared__ float sred[384];
  const int tid = threadIdx.x, lane = tid & 31, wv = tid >> 5;

  for (int idx = tid; idx < 6 * 1024; idx += 128) {
    int k = idx >> 10, t = (idx >> 9) & 1, ln = (idx >> 4) & 31, i = idx & 15;
    W2B[idx] = (_Float16)rW2[k * 1024 + ((ln & 16) + i) * 32 + t * 16 + (ln & 15)];
  }
  for (int idx = tid; idx < 384; idx += 128) sred[idx] = 0.f;
  const int ebase = blockIdx.x * 64 + wv * 16;
  if (lane < 16) {
    int e = ebase + lane;
    ewr[wv][lane][0] = w[e];
    ewr[wv][lane][1] = uR[e * 4 + 3];
  }
  __syncthreads();

  const int e = lane & 15, jb = (lane >> 4) * 16, col0 = lane & 15;
  const float wE = ewr[wv][e][0], rE = ewr[wv][e][1];

  for (int k = 0; k < 6; ++k) {
#pragma unroll
    for (int j2 = 0; j2 < 16; ++j2) {
      int j = jb + j2;
      float h = fmaxf(wE * bn1[k * 32 + j] + rE * bn1[192 + k * 32 + j] +
                          bn1[384 + k * 32 + j], 0.f);
      stage[wv][e][j] = (_Float16)h;
    }
    __syncthreads();
    v16h a = loadA(&stage[wv][0][0], lane);
#pragma unroll
    for (int t = 0; t < 2; ++t) {
      v16h b = *reinterpret_cast<const v16h*>(&W2B[((k * 2 + t) * 32 + lane) * 16]);
      float bias = rb2[k * 32 + t * 16 + col0];
      v8f acc = {bias, bias, bias, bias, bias, bias, bias, bias};
      acc = wmma_f16(a, b, acc);
      float s1 = 0.f, s2 = 0.f;
#pragma unroll
      for (int j = 0; j < 8; ++j) { float y = acc[j]; s1 += y; s2 += y * y; }
      atomicAdd(&sred[(k * 32 + t * 16 + col0) * 2 + 0], s1);
      atomicAdd(&sred[(k * 32 + t * 16 + col0) * 2 + 1], s2);
    }
    __syncthreads();
  }
  for (int idx = tid; idx < 384; idx += 128) atomicAdd(&stats2[idx], sred[idx]);
}

// ---------------------------------------------------------------------------
// K4: fold BN2 into per-(k,col) scale/shift
// ---------------------------------------------------------------------------
__global__ void k_bn2(const float* __restrict__ stats2, const float* __restrict__ rg2,
                      const float* __restrict__ rbe2, float* __restrict__ bn2) {
  int t = threadIdx.x;
  if (t >= 192) return;
  int k = t >> 5, j = t & 31;
  const float invE = 1.f / (float)EE;
  float mu  = stats2[(k * 32 + j) * 2 + 0] * invE;
  float ms  = stats2[(k * 32 + j) * 2 + 1] * invE;
  float var = ms - mu * mu;
  float s   = rsqrtf(var + 1e-5f) * rg2[k * 32 + j];
  bn2[(k * 32 + j) * 2 + 0] = s;
  bn2[(k * 32 + j) * 2 + 1] = rbe2[k * 32 + j] - mu * s;
}

// ---------------------------------------------------------------------------
// K5 (WMMA): conv A. Per wave of 16 edges:
//  h1 -> (WMMA) y -> BN2+relu -> H -> (WMMA) R projections (a00,a01,a10,a11)
//  then equivariant messages m0[16], m1[16][3] scattered via atomics.
// ---------------------------------------------------------------------------
__global__ __launch_bounds__(128) void k_convA(
    const int* __restrict__ src, const int* __restrict__ dst,
    const float* __restrict__ cIn, const float* __restrict__ vIn,
    const float* __restrict__ w, const float* __restrict__ uR,
    const float* __restrict__ bn1, const float* __restrict__ rW2,
    const float* __restrict__ rb2, const float* __restrict__ bn2,
    const float* __restrict__ w3a00, const float* __restrict__ b3a00,
    const float* __restrict__ w3a01, const float* __restrict__ b3a01,
    const float* __restrict__ w3a10, const float* __restrict__ b3a10,
    const float* __restrict__ w3a11, const float* __restrict__ b3a11,
    float* __restrict__ h0acc, float* __restrict__ h1acc) {
  __shared__ alignas(32) _Float16 W2B[4 * 2 * 512];
  __shared__ alignas(32) _Float16 W3B[6 * 512];
  __shared__ alignas(32) _Float16 stage[4][16][32];
  __shared__ float Rall[4][16][96];   // cols: a00 0-15, a01 16-31, a10 32-47, a11 48-95
  __shared__ float esc[4][16][16];    // w,r,A0,A1,B1[3],p[3][3]
  __shared__ int   edst[4][16];
  const int tid = threadIdx.x, lane = tid & 31, wv = tid >> 5;
  const int col0 = lane & 15;
  const int ebase = blockIdx.x * 64 + wv * 16;

  for (int idx = tid; idx < 4 * 1024; idx += 128) {
    int k = idx >> 10, t = (idx >> 9) & 1, ln = (idx >> 4) & 31, i = idx & 15;
    W2B[idx] = (_Float16)rW2[k * 1024 + ((ln & 16) + i) * 32 + t * 16 + (ln & 15)];
  }
  stageW3(W3B + 0 * 512, w3a00, 16, 0, tid);
  stageW3(W3B + 1 * 512, w3a01, 16, 0, tid);
  stageW3(W3B + 2 * 512, w3a10, 16, 0, tid);
  stageW3(W3B + 3 * 512, w3a11, 48, 0, tid);
  stageW3(W3B + 4 * 512, w3a11, 48, 16, tid);
  stageW3(W3B + 5 * 512, w3a11, 48, 32, tid);

  if (lane < 16) {
    int e = ebase + lane;
    int s = src[e], d = dst[e];
    float ux = uR[e * 4 + 0], uy = uR[e * 4 + 1], uz = uR[e * 4 + 2], r = uR[e * 4 + 3];
    const float c1 = 0.48860251190292f;
    float Y10 = c1 * uy, Y11 = c1 * uz, Y12 = c1 * ux;   // Y1 = c1*(y,z,x)
    float cs = cIn[s];
    float v0 = vIn[s * 3 + 0], v1 = vIn[s * 3 + 1], v2 = vIn[s * 3 + 2];
    const float Y0 = 0.28209479177387814f;
    float Y20 = 1.0925484305920792f * ux * uy;
    float Y21 = 1.0925484305920792f * uy * uz;
    float Y22 = 0.31539156525252005f * (3.f * uz * uz - 1.f);
    float Y23 = 1.0925484305920792f * ux * uz;
    float Y24 = 0.5462742152960396f * (ux * ux - uy * uy);
    const float s2c = 0.7071067811865476f, s6c = 0.4082482904638630f,
                s3c = 0.5773502691896258f;
    float k0d = Y0 * s3c;
    // p_f = K_f @ v_src
    float p00 = k0d * v0, p01 = k0d * v1, p02 = k0d * v2;
    float p10 = s2c * ( Y12 * v1 - Y11 * v2);
    float p11 = s2c * (-Y12 * v0 + Y10 * v2);
    float p12 = s2c * ( Y11 * v0 - Y10 * v1);
    float K200 = -s6c * Y22 - s2c * Y24, K201 = s2c * Y21, K202 = s2c * Y20;
    float K211 = 2.f * s6c * Y22, K212 = s2c * Y23;
    float K222 = -s6c * Y22 + s2c * Y24;
    float p20 = K200 * v0 + K201 * v1 + K202 * v2;
    float p21 = K201 * v0 + K211 * v1 + K212 * v2;
    float p22 = K202 * v0 + K212 * v1 + K222 * v2;
    esc[wv][lane][0]  = w[e];
    esc[wv][lane][1]  = r;
    esc[wv][lane][2]  = Y0 * cs;                          // A0
    esc[wv][lane][3]  = Y10 * v0 + Y11 * v1 + Y12 * v2;   // A1 = Y1.v
    esc[wv][lane][4]  = Y10 * cs;                         // B1[0..2] = Y1*c
    esc[wv][lane][5]  = Y11 * cs;
    esc[wv][lane][6]  = Y12 * cs;
    esc[wv][lane][7]  = p00; esc[wv][lane][8]  = p01; esc[wv][lane][9]  = p02;
    esc[wv][lane][10] = p10; esc[wv][lane][11] = p11; esc[wv][lane][12] = p12;
    esc[wv][lane][13] = p20; esc[wv][lane][14] = p21; esc[wv][lane][15] = p22;
    edst[wv][lane] = d;
  }
  __syncthreads();

  const int e = lane & 15, jb = (lane >> 4) * 16;
  const float wE = esc[wv][e][0], rE = esc[wv][e][1];

  for (int k = 0; k < 4; ++k) {
#pragma unroll
    for (int j2 = 0; j2 < 16; ++j2) {
      int j = jb + j2;
      float h = fmaxf(wE * bn1[k * 32 + j] + rE * bn1[192 + k * 32 + j] +
                          bn1[384 + k * 32 + j], 0.f);
      stage[wv][e][j] = (_Float16)h;
    }
    __syncthreads();
    v16h a = loadA(&stage[wv][0][0], lane);
#pragma unroll
    for (int t = 0; t < 2; ++t) {
      v16h b = *reinterpret_cast<const v16h*>(&W2B[((k * 2 + t) * 32 + lane) * 16]);
      float bias = rb2[k * 32 + t * 16 + col0];
      v8f acc = {bias, bias, bias, bias, bias, bias, bias, bias};
      acc = wmma_f16(a, b, acc);
      float sc = bn2[(k * 32 + t * 16 + col0) * 2 + 0];
      float sh = bn2[(k * 32 + t * 16 + col0) * 2 + 1];
#pragma unroll
      for (int j = 0; j < 8; ++j) {
        float H = fmaxf(acc[j] * sc + sh, 0.f);
        stage[wv][j + 8 * (lane >> 4)][t * 16 + col0] = (_Float16)H;
      }
    }
    __syncthreads();
    v16h ah = loadA(&stage[wv][0][0], lane);
    const int nt = (k == 3) ? 3 : 1;
    const int co = k * 16;
    const float* bp = (k == 0) ? b3a00 : (k == 1) ? b3a01 : (k == 2) ? b3a10 : b3a11;
    for (int tt = 0; tt < nt; ++tt) {
      v16h b = *reinterpret_cast<const v16h*>(&W3B[((k + tt) * 32 + lane) * 16]);
      float bias = bp[tt * 16 + col0];
      v8f acc = {bias, bias, bias, bias, bias, bias, bias, bias};
      acc = wmma_f16(ah, b, acc);
#pragma unroll
      for (int j = 0; j < 8; ++j)
        Rall[wv][j + 8 * (lane >> 4)][co + tt * 16 + col0] = acc[j];
    }
    __syncthreads();
  }

  // message scatter: lane handles edge (lane&15), channels ob..ob+7
  const int e2 = lane & 15, ob = (lane >> 4) * 8;
  const int dn = edst[wv][e2];
  const float A0 = esc[wv][e2][2], A1 = esc[wv][e2][3];
  float B1d[3] = {esc[wv][e2][4], esc[wv][e2][5], esc[wv][e2][6]};
  float p[3][3];
#pragma unroll
  for (int f = 0; f < 3; ++f)
#pragma unroll
    for (int d = 0; d < 3; ++d) p[f][d] = esc[wv][e2][7 + f * 3 + d];
#pragma unroll
  for (int oo = 0; oo < 8; ++oo) {
    int o = ob + oo;
    float m0 = Rall[wv][e2][o] * A0 + Rall[wv][e2][32 + o] * A1;
    atomicAdd(&h0acc[dn * 16 + o], m0);
    float ra01 = Rall[wv][e2][16 + o];
    float f0 = Rall[wv][e2][48 + 3 * o + 0];
    float f1 = Rall[wv][e2][48 + 3 * o + 1];
    float f2 = Rall[wv][e2][48 + 3 * o + 2];
#pragma unroll
    for (int d = 0; d < 3; ++d) {
      float m1 = ra01 * B1d[d] + f0 * p[0][d] + f1 * p[1][d] + f2 * p[2][d];
      atomicAdd(&h1acc[(dn * 16 + o) * 3 + d], m1);
    }
  }
}

// ---------------------------------------------------------------------------
// K6: node update A: mean-aggregate + self-interaction + norm_bias
// ---------------------------------------------------------------------------
__global__ void k_nodeA(const float* __restrict__ cIn, const float* __restrict__ vIn,
                        const float* __restrict__ selfA0, const float* __restrict__ selfA1,
                        const float* __restrict__ nbias0, const float* __restrict__ nbias1,
                        const float* __restrict__ deg, const float* __restrict__ h0acc,
                        const float* __restrict__ h1acc, float* __restrict__ h0n,
                        float* __restrict__ h1n) {
  int t = blockIdx.x * blockDim.x + threadIdx.x;
  if (t >= NN * 16) return;
  int n = t >> 4, o = t & 15;
  float dg = deg[n];
  float inv = 1.f / fmaxf(dg, 1.f);
  float mask = (dg > 0.f) ? 1.f : 0.f;
  float h0 = h0acc[t] * inv + selfA0[o] * cIn[n] * mask;
  float nr = sqrtf(h0 * h0 + 1e-12f);
  h0n[t] = fmaxf(nr + nbias0[o], 0.f) * (h0 / fmaxf(nr, 1e-8f));
  float a  = h1acc[t * 3 + 0] * inv + selfA1[o] * vIn[n * 3 + 0] * mask;
  float b  = h1acc[t * 3 + 1] * inv + selfA1[o] * vIn[n * 3 + 1] * mask;
  float c2 = h1acc[t * 3 + 2] * inv + selfA1[o] * vIn[n * 3 + 2] * mask;
  nr = sqrtf(a * a + b * b + c2 * c2 + 1e-12f);
  float sc = fmaxf(nr + nbias1[o], 0.f) / fmaxf(nr, 1e-8f);
  h1n[t * 3 + 0] = sc * a;
  h1n[t * 3 + 1] = sc * b;
  h1n[t * 3 + 2] = sc * c2;
}

// ---------------------------------------------------------------------------
// K7 (WMMA): conv B. Per wave of 16 edges: H4,H5 via WMMA, Rb01/Rb11
// projections via WMMA, then mB = kb01@h0src + kb11@h1src, scatter atomics.
// ---------------------------------------------------------------------------
__global__ __launch_bounds__(128) void k_convB(
    const int* __restrict__ src, const int* __restrict__ dst,
    const float* __restrict__ w, const float* __restrict__ uR,
    const float* __restrict__ bn1, const float* __restrict__ rW2,
    const float* __restrict__ rb2, const float* __restrict__ bn2,
    const float* __restrict__ w3b01, const float* __restrict__ b3b01,
    const float* __restrict__ w3b11, const float* __restrict__ b3b11,
    const float* __restrict__ h0n, const float* __restrict__ h1n,
    float* __restrict__ outacc) {
  __shared__ alignas(32) _Float16 W2B[2 * 2 * 512];
  __shared__ alignas(32) _Float16 W3B[8 * 512];
  __shared__ alignas(32) _Float16 stage[4][16][32];
  __shared__ float Rb[4][16][128];    // cols: b01 0-31, b11 32-127
  __shared__ float escF[4][16][5];    // w, r, ux, uy, uz
  __shared__ int   escI[4][16][2];    // src, dst
  const int tid = threadIdx.x, lane = tid & 31, wv = tid >> 5;
  const int col0 = lane & 15;
  const int ebase = blockIdx.x * 64 + wv * 16;

  for (int idx = tid; idx < 2 * 1024; idx += 128) {
    int kk = idx >> 10, t = (idx >> 9) & 1, ln = (idx >> 4) & 31, i = idx & 15;
    W2B[idx] = (_Float16)rW2[(4 + kk) * 1024 + ((ln & 16) + i) * 32 + t * 16 + (ln & 15)];
  }
  stageW3(W3B + 0 * 512, w3b01, 32, 0, tid);
  stageW3(W3B + 1 * 512, w3b01, 32, 16, tid);
  for (int tt = 0; tt < 6; ++tt)
    stageW3(W3B + (2 + tt) * 512, w3b11, 96, tt * 16, tid);

  if (lane < 16) {
    int e = ebase + lane;
    escF[wv][lane][0] = w[e];
    escF[wv][lane][1] = uR[e * 4 + 3];
    escF[wv][lane][2] = uR[e * 4 + 0];
    escF[wv][lane][3] = uR[e * 4 + 1];
    escF[wv][lane][4] = uR[e * 4 + 2];
    escI[wv][lane][0] = src[e];
    escI[wv][lane][1] = dst[e];
  }
  __syncthreads();

  const int e = lane & 15, jb = (lane >> 4) * 16;
  const float wE = escF[wv][e][0], rE = escF[wv][e][1];

  for (int kk = 0; kk < 2; ++kk) {
    const int k = 4 + kk;
#pragma unroll
    for (int j2 = 0; j2 < 16; ++j2) {
      int j = jb + j2;
      float h = fmaxf(wE * bn1[k * 32 + j] + rE * bn1[192 + k * 32 + j] +
                          bn1[384 + k * 32 + j], 0.f);
      stage[wv][e][j] = (_Float16)h;
    }
    __syncthreads();
    v16h a = loadA(&stage[wv][0][0], lane);
#pragma unroll
    for (int t = 0; t < 2; ++t) {
      v16h b = *reinterpret_cast<const v16h*>(&W2B[((kk * 2 + t) * 32 + lane) * 16]);
      float bias = rb2[k * 32 + t * 16 + col0];
      v8f acc = {bias, bias, bias, bias, bias, bias, bias, bias};
      acc = wmma_f16(a, b, acc);
      float sc = bn2[(k * 32 + t * 16 + col0) * 2 + 0];
      float sh = bn2[(k * 32 + t * 16 + col0) * 2 + 1];
#pragma unroll
      for (int j = 0; j < 8; ++j) {
        float H = fmaxf(acc[j] * sc + sh, 0.f);
        stage[wv][j + 8 * (lane >> 4)][t * 16 + col0] = (_Float16)H;
      }
    }
    __syncthreads();
    v16h ah = loadA(&stage[wv][0][0], lane);
    const int nt = kk ? 6 : 2;
    const int tb = kk ? 2 : 0;
    const int co = kk ? 32 : 0;
    const float* bp = kk ? b3b11 : b3b01;
    for (int tt = 0; tt < nt; ++tt) {
      v16h b = *reinterpret_cast<const v16h*>(&W3B[((tb + tt) * 32 + lane) * 16]);
      float bias = bp[tt * 16 + col0];
      v8f acc = {bias, bias, bias, bias, bias, bias, bias, bias};
      acc = wmma_f16(ah, b, acc);
#pragma unroll
      for (int j = 0; j < 8; ++j)
        Rb[wv][j + 8 * (lane >> 4)][co + tt * 16 + col0] = acc[j];
    }
    __syncthreads();
  }

  // compute phase: lane handles edge (lane&15), output block o2=(lane>>4)
  const int e2 = lane & 15, o2 = lane >> 4;
  const int s = escI[wv][e2][0], dn = escI[wv][e2][1];
  float ux = escF[wv][e2][2], uy = escF[wv][e2][3], uz = escF[wv][e2][4];
  const float c1 = 0.48860251190292f;
  float Y10 = c1 * uy, Y11 = c1 * uz, Y12 = c1 * ux;
  const float Y0 = 0.28209479177387814f;
  float Y20 = 1.0925484305920792f * ux * uy;
  float Y21 = 1.0925484305920792f * uy * uz;
  float Y22 = 0.31539156525252005f * (3.f * uz * uz - 1.f);
  float Y23 = 1.0925484305920792f * ux * uz;
  float Y24 = 0.5462742152960396f * (ux * ux - uy * uy);
  const float s2c = 0.7071067811865476f, s6c = 0.4082482904638630f,
              s3c = 0.5773502691896258f;
  float k0d = Y0 * s3c;
  float K200 = -s6c * Y22 - s2c * Y24, K201 = s2c * Y21, K202 = s2c * Y20;
  float K211 = 2.f * s6c * Y22, K212 = s2c * Y23;
  float K222 = -s6c * Y22 + s2c * Y24;

  float dot01 = 0.f;
#pragma unroll
  for (int i = 0; i < 16; ++i)
    dot01 += Rb[wv][e2][o2 * 16 + i] * h0n[s * 16 + i];

  float g00 = 0, g01 = 0, g02 = 0, g10 = 0, g11 = 0, g12 = 0,
        g20 = 0, g21 = 0, g22 = 0;
#pragma unroll
  for (int i = 0; i < 16; ++i) {
    float hx = h1n[(s * 16 + i) * 3 + 0];
    float hy = h1n[(s * 16 + i) * 3 + 1];
    float hz = h1n[(s * 16 + i) * 3 + 2];
    int rbase = 32 + o2 * 48 + i * 3;
    float c0 = Rb[wv][e2][rbase + 0];
    float cA = Rb[wv][e2][rbase + 1];
    float cB = Rb[wv][e2][rbase + 2];
    g00 += c0 * hx; g01 += c0 * hy; g02 += c0 * hz;
    g10 += cA * hx; g11 += cA * hy; g12 += cA * hz;
    g20 += cB * hx; g21 += cB * hy; g22 += cB * hz;
  }
  float mb0 = k0d * g00 + s2c * ( Y12 * g11 - Y11 * g12) +
              (K200 * g20 + K201 * g21 + K202 * g22) + dot01 * Y10;
  float mb1 = k0d * g01 + s2c * (-Y12 * g10 + Y10 * g12) +
              (K201 * g20 + K211 * g21 + K212 * g22) + dot01 * Y11;
  float mb2 = k0d * g02 + s2c * ( Y11 * g10 - Y10 * g11) +
              (K202 * g20 + K212 * g21 + K222 * g22) + dot01 * Y12;
  atomicAdd(&outacc[dn * 6 + o2 * 3 + 0], mb0);
  atomicAdd(&outacc[dn * 6 + o2 * 3 + 1], mb1);
  atomicAdd(&outacc[dn * 6 + o2 * 3 + 2], mb2);
}

// ---------------------------------------------------------------------------
// K8: node update B -> final output [N,2,3]
// ---------------------------------------------------------------------------
__global__ void k_nodeB(const float* __restrict__ selfB1, const float* __restrict__ deg,
                        const float* __restrict__ outacc, const float* __restrict__ h1n,
                        float* __restrict__ out) {
  int n = blockIdx.x * blockDim.x + threadIdx.x;
  if (n >= NN) return;
  float dg = deg[n];
  float inv = 1.f / fmaxf(dg, 1.f);
  float mask = (dg > 0.f) ? 1.f : 0.f;
#pragma unroll
  for (int o2 = 0; o2 < 2; ++o2)
#pragma unroll
    for (int d = 0; d < 3; ++d) {
      float s = 0.f;
#pragma unroll
      for (int i = 0; i < 16; ++i)
        s += selfB1[o2 * 16 + i] * h1n[(n * 16 + i) * 3 + d];
      out[n * 6 + o2 * 3 + d] = outacc[n * 6 + o2 * 3 + d] * inv + s * mask;
    }
}

// ---------------------------------------------------------------------------
// launcher
// ---------------------------------------------------------------------------
extern "C" void kernel_launch(void* const* d_in, const int* in_sizes, int n_in,
                              void* d_out, int out_size, void* d_ws, size_t ws_size,
                              hipStream_t stream) {
  (void)in_sizes; (void)n_in; (void)out_size; (void)ws_size;
  const int*   src   = (const int*)  d_in[0];
  const int*   dst   = (const int*)  d_in[1];
  const float* pos   = (const float*)d_in[2];
  const float* cIn   = (const float*)d_in[3];
  const float* vIn   = (const float*)d_in[4];
  const float* w     = (const float*)d_in[5];
  const float* rW1   = (const float*)d_in[6];
  const float* rb1   = (const float*)d_in[7];
  const float* rg1   = (const float*)d_in[8];
  const float* rbe1  = (const float*)d_in[9];
  const float* rW2   = (const float*)d_in[10];
  const float* rb2   = (const float*)d_in[11];
  const float* rg2   = (const float*)d_in[12];
  const float* rbe2  = (const float*)d_in[13];
  const float* w3a00 = (const float*)d_in[14]; const float* b3a00 = (const float*)d_in[15];
  const float* w3a01 = (const float*)d_in[16]; const float* b3a01 = (const float*)d_in[17];
  const float* w3a10 = (const float*)d_in[18]; const float* b3a10 = (const float*)d_in[19];
  const float* w3a11 = (const float*)d_in[20]; const float* b3a11 = (const float*)d_in[21];
  const float* w3b01 = (const float*)d_in[22]; const float* b3b01 = (const float*)d_in[23];
  const float* w3b11 = (const float*)d_in[24]; const float* b3b11 = (const float*)d_in[25];
  const float* selfA0 = (const float*)d_in[26];
  const float* selfA1 = (const float*)d_in[27];
  const float* selfB1 = (const float*)d_in[28];
  const float* nbias0 = (const float*)d_in[29];
  const float* nbias1 = (const float*)d_in[30];

  float* ws     = (float*)d_ws;
  float* uR     = ws + OFF_UR;
  float* deg    = ws + OFF_DEG;
  float* h0acc  = ws + OFF_H0A;
  float* h1acc  = ws + OFF_H1A;
  float* outacc = ws + OFF_OUTA;
  float* stats1 = ws + OFF_S1;
  float* stats2 = ws + OFF_S2;
  float* bn1    = ws + OFF_BN1;
  float* bn2    = ws + OFF_BN2;
  float* h0n    = ws + OFF_H0N;
  float* h1n    = ws + OFF_H1N;

  k_zero<<<(ZERO_CNT + 255) / 256, 256, 0, stream>>>(ws + ZERO_OFF, ZERO_CNT);
  k_geom<<<EE / 256, 256, 0, stream>>>(src, dst, pos, w, uR, deg, stats1);
  k_bn1<<<1, 192, 0, stream>>>(stats1, rW1, rb1, rg1, rbe1, bn1);
  k_stats2<<<EE / 64, 128, 0, stream>>>(w, uR, bn1, rW2, rb2, stats2);
  k_bn2<<<1, 192, 0, stream>>>(stats2, rg2, rbe2, bn2);
  k_convA<<<EE / 64, 128, 0, stream>>>(src, dst, cIn, vIn, w, uR, bn1, rW2, rb2, bn2,
                                       w3a00, b3a00, w3a01, b3a01, w3a10, b3a10,
                                       w3a11, b3a11, h0acc, h1acc);
  k_nodeA<<<(NN * 16 + 255) / 256, 256, 0, stream>>>(cIn, vIn, selfA0, selfA1, nbias0,
                                                     nbias1, deg, h0acc, h1acc, h0n, h1n);
  k_convB<<<EE / 64, 128, 0, stream>>>(src, dst, w, uR, bn1, rW2, rb2, bn2,
                                       w3b01, b3b01, w3b11, b3b11, h0n, h1n, outacc);
  k_nodeB<<<(NN + 255) / 256, 256, 0, stream>>>(selfB1, deg, outacc, h1n, (float*)d_out);
}
